// ConvNN_71820443124000
// MI455X (gfx1250) — matrix operand order repaired and checked by reading, not compiled, not run on hardware
//
#include <hip/hip_runtime.h>
#include <stdint.h>

// ---------------------------------------------------------------------------
// Problem constants (from reference): B=4, CIN=COUT=128, K=9, NS=NIN=16384, H=64
// ---------------------------------------------------------------------------
#define BB      4
#define CIN     128
#define COUT    128
#define KK      9
#define NS      16384
#define NIN     16384
#define HH      64
#define W0      30.0f

// main-kernel tiling
#define TN      8                 // n-values per workgroup
#define ROWS    (BB * TN)         // 32 M-rows = (b,n) pairs
#define CI_P    130               // xg LDS pitch (pad: 1-dword bank rotation)
#define ZP      136               // z LDS pitch  (pad: 4-dword bank rotation)
#define WP      136               // W3 LDS pitch (produced by TDM pad feature)

typedef __attribute__((ext_vector_type(16))) __bf16   bf16x16;
typedef __attribute__((ext_vector_type(8)))  float    f32x8;
typedef __attribute__((ext_vector_type(8)))  uint32_t u32x8;
typedef __attribute__((ext_vector_type(4)))  uint32_t u32x4;
typedef __attribute__((ext_vector_type(8)))  int32_t  i32x8;
typedef __attribute__((ext_vector_type(4)))  int32_t  i32x4;

__device__ __forceinline__ uint16_t f2bf(float f) {   // round-to-nearest-even
  union { float f; uint32_t u; } v; v.f = f;
  uint32_t u = v.u;
  return (uint16_t)((u + 0x7FFFu + ((u >> 16) & 1u)) >> 16);
}
__device__ __forceinline__ float bf2f(uint16_t h) {
  union { float f; uint32_t u; } v; v.u = ((uint32_t)h) << 16; return v.f;
}

// ---------------------------------------------------------------------------
// Kernel 1: SIREN MLP  h2[kn,h] = sin( sin(W0*(coords@W1+b1)) @ W2 + b2 )
// 4 rows per 256-thread block, 64 threads per row. Trivial cost (~0.6 GFLOP).
// ---------------------------------------------------------------------------
__global__ void __launch_bounds__(256)
mlp_kernel(const float* __restrict__ coords, const float* __restrict__ W1,
           const float* __restrict__ b1, const float* __restrict__ W2,
           const float* __restrict__ b2, float* __restrict__ h2out)
{
  __shared__ float h1s[4][HH];
  const int t  = threadIdx.x;
  const int rl = t >> 6;            // row within block
  const int hh = t & 63;
  const long kn = (long)blockIdx.x * 4 + rl;
  const float c0 = coords[kn * 2 + 0];
  const float c1 = coords[kn * 2 + 1];
  float p = fmaf(c0, W1[hh], fmaf(c1, W1[HH + hh], b1[hh]));
  h1s[rl][hh] = __sinf(W0 * p);
  __syncthreads();
  float acc = b2[hh];
#pragma unroll 8
  for (int k2 = 0; k2 < HH; ++k2)
    acc = fmaf(h1s[rl][k2], W2[k2 * HH + hh], acc);
  h2out[kn * HH + hh] = __sinf(acc);
}

// ---------------------------------------------------------------------------
// Kernel 1b: W3b[h][co][ci] = bf16(W3[h, co*CIN+ci]) for h<64,
//            W3b[64][co][ci] = bf16(b3[co*CIN+ci])   (b3 folded as 65th K-step)
// W3's native layout already matches [h][co][ci] — pure convert.
// ---------------------------------------------------------------------------
__global__ void __launch_bounds__(256)
cvt_w3_kernel(const float* __restrict__ W3, const float* __restrict__ b3,
              uint16_t* __restrict__ W3b)
{
  const int i = blockIdx.x * 256 + threadIdx.x;
  if (i >= 65 * COUT * CIN) return;
  const float v = (i < HH * COUT * CIN) ? W3[i] : b3[i - HH * COUT * CIN];
  W3b[i] = f2bf(v);
}

// ---------------------------------------------------------------------------
// Kernel 2 (main): per workgroup, an n-tile of TN=8.
//   gather x (L2-resident) -> bf16 LDS
//   for h in 0..64:
//     wave 0 issues a TDM tensor_load_to_lds for the 32KB W3 slice (with the
//       pad feature inserting 16B every 256B -> LDS pitch 136 for conflict-free
//       B-fragment reads), overlapped with...
//     z[(b,n),ci] = sum_kk h2[kk,n,h] * x_g        (VALU, co-executes w/ WMMA)
//     s_wait_tensorcnt 0 ; barrier
//     out[(b,n),co] += z @ W3b[h]                  (v_wmma_f32_16x16x32_bf16)
//   epilogue: out[b,co,n] = acc + bias[co]
// 512 threads = 16 wave32s; wave grid 2(M) x 8(N); each wave owns one
// 16x16 fp32 accumulator tile (8 VGPRs) over K = 65*128 = 8320.
// ---------------------------------------------------------------------------
__global__ void __launch_bounds__(512)
conv_main_kernel(const float* __restrict__ x, const int* __restrict__ nbrs,
                 const float* __restrict__ h2g, const uint16_t* __restrict__ W3b,
                 const float* __restrict__ bias, float* __restrict__ out)
{
  __shared__ __attribute__((aligned(16))) int      nbrS[KK * TN];            // 288 B
  __shared__ __attribute__((aligned(16))) uint16_t xgS[KK * BB * TN * CI_P]; // 74.9 KB
  __shared__ __attribute__((aligned(16))) float    h2tS[KK * TN * HH];       // 18.4 KB
  __shared__ __attribute__((aligned(16))) uint16_t wtS[COUT * WP];           // 34.8 KB
  __shared__ __attribute__((aligned(16))) uint16_t zS[ROWS * ZP];            // 8.7 KB

  const int t  = threadIdx.x;
  const int n0 = blockIdx.x * TN;

  // ---- neighbour indices for this tile: nbrs[kk, n0+nl] ----
  if (t < KK * TN) nbrS[t] = nbrs[(t / TN) * NS + n0 + (t % TN)];

  // ---- h2 tile: h2tS[(kk*TN+nl)*64 + h] ----
  for (int i = t; i < KK * TN * HH; i += 512) {
    const int row = i >> 6;                 // kk*TN + nl
    const int kk  = row / TN, nl = row % TN;
    h2tS[i] = h2g[((long)(kk * NS + n0 + nl)) * HH + (i & 63)];
  }
  __syncthreads();

  // ---- gather x[b,ci,nbr[kk,n]] -> bf16 LDS (x is L2-resident: 32MB < 192MB) ----
  for (int e = t; e < KK * BB * CIN * TN; e += 512) {
    const int nl = e & (TN - 1);
    const int r  = e >> 3;                  // (kk*4+b)*128 + ci
    const int ci = r & 127;
    const int bb = (r >> 7) & 3;
    const int kk = r >> 9;
    const int idx = nbrS[kk * TN + nl];
    const float xv = x[((long)(bb * CIN + ci)) * NIN + idx];
    xgS[((kk * BB + bb) * TN + nl) * CI_P + ci] = f2bf(xv);
  }
  __syncthreads();

  // ---- wave identity: 2 M-tiles x 8 N-tiles ----
  const int lane = t & 31;
  const int wv   = t >> 5;
  const int mt   = wv & 1;                  // M-tile 0..1
  const int nt   = wv >> 1;                 // N-tile 0..7
  f32x8 acc = {0.f, 0.f, 0.f, 0.f, 0.f, 0.f, 0.f, 0.f};

  // z-generation assignment: thread -> (row rz, 8 consecutive ci)
  const int rz    = t & 31;                 // row = nl*4 + b
  const int bb_z  = rz & 3;
  const int nl_z  = rz >> 2;
  const int ci0_z = (t >> 5) * 8;

  // LDS byte offset of the W3 staging buffer (generic LDS address: offset in
  // the low 32 bits, per the ISA aperture mapping).
  const uint32_t lds_wt = (uint32_t)(size_t)(void*)wtS;

  for (int h = 0; h < HH + 1; ++h) {        // 65 K-steps (last one = b3 term)
    // -- TDM: DMA this h's W3 slice [128co x 128ci] bf16 -> LDS, pitch 136.
    //    Issued by wave 0 only (tensor ops ignore EXEC; one issue per WG).
    //    Previous iteration's trailing barrier guarantees wtS is no longer
    //    being read when the DMA lands.
    if (wv == 0) {
      const uint64_t ga = (uint64_t)(size_t)(W3b + (size_t)h * (COUT * CIN));
      u32x4 g0;
      g0.x = 1u;                                  // count=1, user load descriptor
      g0.y = lds_wt;                              // D#.lds_addr
      g0.z = (uint32_t)ga;                        // D#.global_addr[31:0]
      g0.w = (uint32_t)(ga >> 32) | (2u << 30);   // global_addr[56:32] | type=2
      i32x8 g1;
      g1[0] = (int)((1u << 16)   |                // data_size = 2 bytes
                    (1u << 20)   |                // pad_enable
                    (5u << 22)   |                // pad_interval: 64 dwords (256B)
                    (3u << 25));                  // pad_amount:   4 dwords (16B)
      g1[1] = (int)(((uint32_t)(COUT * CIN) & 0xFFFFu) << 16); // tensor_dim0 lo16
      g1[2] = (int)(1u << 16);                    // tensor_dim0 hi = 0, tensor_dim1 = 1
      g1[3] = (int)(((uint32_t)(COUT * CIN) & 0xFFFFu) << 16); // tile_dim0 = 16384
      g1[4] = 1;                                  // tile_dim1 = 1, tile_dim2 = 0
      g1[5] = COUT * CIN;                         // tensor_dim0_stride lo32
      g1[6] = 0;
      g1[7] = 0;
      const i32x4 gz = {0, 0, 0, 0};              // groups 2/3 unused (<=2D tensor)
#if __clang_major__ >= 23
      const i32x8 gz8 = {0, 0, 0, 0, 0, 0, 0, 0};
      __builtin_amdgcn_tensor_load_to_lds(g0, g1, gz, gz, gz8, 0);
#else
      __builtin_amdgcn_tensor_load_to_lds(g0, g1, gz, gz, 0);
#endif
    }

    // -- z chunk (overlaps the TDM DMA):
    //    z[rz, ci0..ci0+7] = sum_kk c_kk * x_g  (fp32 acc, bf16 store)
    float zacc[8] = {0.f, 0.f, 0.f, 0.f, 0.f, 0.f, 0.f, 0.f};
#pragma unroll
    for (int kk = 0; kk < KK; ++kk) {
      const float c = (h < HH) ? h2tS[(kk * TN + nl_z) * HH + h] : 1.0f;
      const uint16_t* xr = &xgS[((kk * BB + bb_z) * TN + nl_z) * CI_P + ci0_z];
#pragma unroll
      for (int j = 0; j < 8; ++j) zacc[j] = fmaf(c, bf2f(xr[j]), zacc[j]);
    }
#pragma unroll
    for (int j = 0; j < 4; ++j) {
      const uint32_t pk = (uint32_t)f2bf(zacc[2 * j]) |
                          ((uint32_t)f2bf(zacc[2 * j + 1]) << 16);
      *(uint32_t*)&zS[rz * ZP + ci0_z + 2 * j] = pk;
    }

    // -- complete the DMA (wave 0), then rendezvous before WMMA reads wtS/zS
    if (wv == 0) __builtin_amdgcn_s_wait_tensorcnt(0);
    __syncthreads();

    // -- WMMA: acc += A(z 16x32) x B(W3 32x16), K=128 in 4 steps
#pragma unroll
    for (int kt = 0; kt < 4; ++kt) {
      // A fragment per ISA 7.12.2 (16-bit A 16x32): lane M=L%16,
      // pair p -> K = 2p + (p>=4 ? 8 : 0) + 8*(L/16)
      u32x8 au;
      {
        const int base = (mt * 16 + (lane & 15)) * ZP + kt * 32 + 8 * (lane >> 4);
#pragma unroll
        for (int p = 0; p < 8; ++p)
          au[p] = *(const uint32_t*)&zS[base + 2 * p + (p >= 4 ? 8 : 0)];
      }
      // B fragment (16-bit B 32x16): lane N=L%16, pair p -> K = 2p + (p>=4 ? 8 : 0) + 8*(L/16);
      // B[k][n] = wtS[co=n][ci=k] so pairs are contiguous in LDS.
      u32x8 bu;
      {
        const int base = (nt * 16 + (lane & 15)) * WP + kt * 32 + 8 * (lane >> 4);
#pragma unroll
        for (int p = 0; p < 8; ++p)
          bu[p] = *(const uint32_t*)&wtS[base + 2 * p + (p >= 4 ? 8 : 0)];
      }
      acc = __builtin_amdgcn_wmma_f32_16x16x32_bf16(
                false, __builtin_bit_cast(bf16x16, au),
                false, __builtin_bit_cast(bf16x16, bu),
                (short)0, acc, false, false);
    }
    __syncthreads();
  }

  // ---- epilogue: C layout (ISA 7.12.2): VGPR r -> M = r + 8*(L/16), N = L%16
  {
    const int co = nt * 16 + (lane & 15);
    const float bv = bias[co];
    const int rbase = mt * 16 + 8 * (lane >> 4);
#pragma unroll
    for (int r = 0; r < 8; ++r) {
      const int row = rbase + r;          // (b,n): b = row%4, nl = row/4
      const int bb = row & 3, nl = row >> 2;
      out[((long)(bb * COUT + co)) * NS + n0 + nl] = acc[r] + bv;
    }
  }
}

// ---------------------------------------------------------------------------
// Launch. Workspace layout (needs >= ~40 MB):
//   [0, 37748736)                : h2 fp32 (147456 x 64)
//   [37748736, +2129920)         : W3b bf16 (65 x 128 x 128)
// ---------------------------------------------------------------------------
extern "C" void kernel_launch(void* const* d_in, const int* in_sizes, int n_in,
                              void* d_out, int out_size, void* d_ws, size_t ws_size,
                              hipStream_t stream) {
  (void)in_sizes; (void)n_in; (void)out_size; (void)ws_size;
  const float* x      = (const float*)d_in[0];
  const int*   nbrs   = (const int*)  d_in[1];
  const float* coords = (const float*)d_in[2];
  const float* W1     = (const float*)d_in[3];
  const float* b1     = (const float*)d_in[4];
  const float* W2     = (const float*)d_in[5];
  const float* b2     = (const float*)d_in[6];
  const float* W3     = (const float*)d_in[7];
  const float* b3     = (const float*)d_in[8];
  const float* bias   = (const float*)d_in[9];
  float* out = (float*)d_out;

  float*    h2ws = (float*)d_ws;
  uint16_t* W3b  = (uint16_t*)((char*)d_ws + (size_t)KK * NS * HH * 4); // 37748736

  mlp_kernel<<<(KK * NS) / 4, 256, 0, stream>>>(coords, W1, b1, W2, b2, h2ws);
  cvt_w3_kernel<<<(65 * COUT * CIN + 255) / 256, 256, 0, stream>>>(W3, b3, W3b);
  conv_main_kernel<<<NS / TN, 512, 0, stream>>>(x, nbrs, h2ws, W3b, bias, out);
}
